// CovariantAttention_54494545051805
// MI455X (gfx1250) — matrix operand order repaired and checked by reading, not compiled, not run on hardware
//
#include <hip/hip_runtime.h>
#include <hip/hip_bf16.h>

// ---------------------------------------------------------------------------
// CovariantAttention on MI455X (gfx1250, wave32, WMMA 16x16x32 bf16)
// B=4, L=2048, D=1024, H=16, hd=64
// ---------------------------------------------------------------------------

typedef __attribute__((ext_vector_type(16))) __bf16 v16bf;
typedef __attribute__((ext_vector_type(8)))  __bf16 v8bf;
typedef __attribute__((ext_vector_type(8)))  float  v8f;

#define WMMA_BF16(A, B, C) \
  __builtin_amdgcn_wmma_f32_16x16x32_bf16(false, (A), false, (B), (short)0, (C), false, false)

// A-fragment (16x32 bf16), row-major source with arbitrary row pointer.
// Contiguous k order used here (B loads match): lane (row = lane&15, hi = lane>>4):
//   elems 0..7  = K(hi*16 .. hi*16+7)     -> 16B contiguous
//   elems 8..15 = K(hi*16+8 .. hi*16+15)  -> 16B contiguous
static __device__ __forceinline__ v16bf load_a_frag(const __bf16* rowp, int hi) {
  v8bf lo = *(const v8bf*)(rowp + hi * 16);
  v8bf hh = *(const v8bf*)(rowp + hi * 16 + 8);
  v16bf a;
#pragma unroll
  for (int i = 0; i < 8; ++i) { a[i] = lo[i]; a[8 + i] = hh[i]; }
  return a;
}

// ---------------------------------------------------------------------------
// Kernel 1a: zc = bf16(concat(z, connection)) [M,2048]; zb = bf16(z) [M,1024]
// ---------------------------------------------------------------------------
__global__ void build_inputs_bf16(const float* __restrict__ z,
                                  const float* __restrict__ con,
                                  __bf16* __restrict__ zc,
                                  __bf16* __restrict__ zb) {
  const int t = blockIdx.x * blockDim.x + threadIdx.x;   // over M*2048
  const int j = t & 2047;
  const int m = t >> 11;
  float v = (j < 1024) ? z[(size_t)m * 1024 + j] : con[(size_t)m * 1024 + (j - 1024)];
  __bf16 b = (__bf16)v;
  zc[t] = b;
  if (j < 1024) zb[(size_t)m * 1024 + j] = b;
}

// ---------------------------------------------------------------------------
// Kernel 1b: dst[n*K + k] = bf16(src[k*N + n])   (weight transpose+convert)
// ---------------------------------------------------------------------------
__global__ void transpose_cvt(const float* __restrict__ src, __bf16* __restrict__ dst,
                              int K, int N) {
  const int t = blockIdx.x * blockDim.x + threadIdx.x;
  if (t >= K * N) return;
  const int k = t / N;
  const int n = t - k * N;
  dst[(size_t)n * K + k] = (__bf16)src[t];
}

// ---------------------------------------------------------------------------
// Kernel 2/4: C[M,N] = A[M,Kd] (bf16 row-major) @ W (bf16, pre-transposed [N,Kd])
// One wave owns a 32(M) x 64(N) tile: two 16-row sub-tiles share the same
// 4 B-fragments -> 8 WMMA per K-step for 6 fragment loads.
// mode 0: f32 row-major Out[M,N]
// mode 1: bf16 head-major   Out[b,h,l,64]   (Q/K for attention)
// mode 2: bf16 head-transp  Out[b,h,64,l]   (V^T for attention)
// ---------------------------------------------------------------------------
__global__ void gemm_bf16_wmma(const __bf16* __restrict__ A, const __bf16* __restrict__ Bt,
                               int M, int Kd, int N, int mode,
                               void* __restrict__ Out, int L, int Hn) {
  const int lane = threadIdx.x & 31;
  const int wave = threadIdx.x >> 5;
  const int ntiles = N >> 6;
  const int wt = blockIdx.x * (blockDim.x >> 5) + wave;
  const int mt = wt / ntiles;
  const int nt = wt - mt * ntiles;
  if (mt >= (M >> 5)) return;
  const int m0 = mt << 5, n0 = nt << 6;
  const int cr = lane & 15, hi = lane >> 4;

  v8f acc[8];
#pragma unroll
  for (int i = 0; i < 8; ++i) acc[i] = (v8f){};

  const __bf16* arow0 = A + (size_t)(m0 + cr) * Kd;
  const __bf16* arow1 = arow0 + (size_t)16 * Kd;
  const __bf16* brow  = Bt + (size_t)(n0 + cr) * Kd;   // lane cr = B column index

  for (int k = 0; k < Kd; k += 32) {
    v16bf a0 = load_a_frag(arow0 + k, hi);
    v16bf a1 = load_a_frag(arow1 + k, hi);
    const __bf16* bb = brow + k + hi * 16;             // 32B contiguous per lane
    v16bf bfr[4];
#pragma unroll
    for (int nb = 0; nb < 4; ++nb)
      bfr[nb] = *(const v16bf*)(bb + (size_t)(16 * nb) * Kd);
    // pull next K-block toward the WGP while the XDL pipe works
    __builtin_prefetch(arow0 + k + 64, 0, 3);
    __builtin_prefetch(bb + 64, 0, 3);
#pragma unroll
    for (int nb = 0; nb < 4; ++nb) {
      acc[nb]     = WMMA_BF16(a0, bfr[nb], acc[nb]);
      acc[4 + nb] = WMMA_BF16(a1, bfr[nb], acc[4 + nb]);
    }
  }

#pragma unroll
  for (int sub = 0; sub < 2; ++sub) {
#pragma unroll
    for (int nb = 0; nb < 4; ++nb) {
      v8f c = acc[sub * 4 + nb];
      const int col = n0 + nb * 16 + cr;
#pragma unroll
      for (int r = 0; r < 8; ++r) {
        const int mrow = m0 + sub * 16 + r + 8 * hi;  // C layout: M = r + 8*(lane>=16)
        const float v = c[r];
        if (mode == 0) {
          ((float*)Out)[(size_t)mrow * N + col] = v;
        } else {
          const int b = mrow / L, l = mrow - b * L;
          const int h = col >> 6, d = col & 63;
          if (mode == 1)
            ((__bf16*)Out)[(((size_t)(b * Hn + h)) * L + l) * 64 + d] = (__bf16)v;
          else
            ((__bf16*)Out)[(((size_t)(b * Hn + h)) * 64 + d) * L + l] = (__bf16)v;
        }
      }
    }
  }
}

// ---------------------------------------------------------------------------
// Kernel 3: streaming sigmoid attention.
// One wave = one (b,h,q0) 32-row query tile (two 16-row sub-tiles that share
// all K/V B-fragments). Sweep K in 32-wide blocks:
//   scores(16x16) via WMMA (K split d=0..31 / 32..63), sigmoid in f32,
//   accumulate row sums, stage P(16x32) bf16 in LDS (C-frag -> A-frag relayout),
//   then 8 WMMA of P x V^T into the two 16x64 output accumulators.
// Epilogue: shfl-reduce row sums across the 16-lane half, divide, store bf16.
// ---------------------------------------------------------------------------
__global__ void attn_sigmoid_wmma(const __bf16* __restrict__ Qh,
                                  const __bf16* __restrict__ Kh,
                                  const __bf16* __restrict__ Vt,
                                  __bf16* __restrict__ att,
                                  const float* __restrict__ tempp,
                                  int L, int Hn, int BH) {
  __shared__ __bf16 pstage[4][2][16 * 32];       // 2KB per wave
  const int lane = threadIdx.x & 31;
  const int wave = threadIdx.x >> 5;
  const int qtiles = L >> 5;                     // 32-row q tiles
  const int wid = blockIdx.x * (blockDim.x >> 5) + wave;
  const int bh = wid / qtiles;
  if (bh >= BH) return;
  const int q0 = (wid - bh * qtiles) << 5;
  const int cr = lane & 15, hi = lane >> 4;
  const float scale = 0.125f * tempp[0];         // hd^-0.5 * temp, hd=64

  const __bf16* Qb = Qh + (size_t)bh * L * 64;
  const __bf16* Kb = Kh + (size_t)bh * L * 64;
  const __bf16* Vb = Vt + (size_t)bh * 64 * L;

  // Q A-fragments for both sub-tiles, d-blocks [0,32) and [32,64)
  const __bf16* qrow0 = Qb + (size_t)(q0 + cr) * 64;
  const __bf16* qrow1 = qrow0 + (size_t)16 * 64;
  v16bf aq00 = load_a_frag(qrow0, hi);
  v16bf aq01 = load_a_frag(qrow0 + 32, hi);
  v16bf aq10 = load_a_frag(qrow1, hi);
  v16bf aq11 = load_a_frag(qrow1 + 32, hi);

  float rs0[8], rs1[8];
#pragma unroll
  for (int r = 0; r < 8; ++r) { rs0[r] = 0.f; rs1[r] = 0.f; }
  v8f o[8];
#pragma unroll
  for (int i = 0; i < 8; ++i) o[i] = (v8f){};

  __bf16* ps0 = &pstage[wave][0][0];
  __bf16* ps1 = &pstage[wave][1][0];

  for (int k32 = 0; k32 < L; k32 += 32) {
#pragma unroll
    for (int kb = 0; kb < 2; ++kb) {
      const int k0 = k32 + kb * 16;
      // B = K^T tile: lane cr holds column cr = K row (k0+cr); contiguous 32B
      const __bf16* kr = Kb + (size_t)(k0 + cr) * 64 + hi * 16;
      v16bf bk0 = *(const v16bf*)(kr);
      v16bf bk1 = *(const v16bf*)(kr + 32);
      v8f s0 = {}, s1 = {};
      s0 = WMMA_BF16(aq00, bk0, s0);
      s0 = WMMA_BF16(aq01, bk1, s0);
      s1 = WMMA_BF16(aq10, bk0, s1);
      s1 = WMMA_BF16(aq11, bk1, s1);
#pragma unroll
      for (int r = 0; r < 8; ++r) {
        const float rel0 = 1.0f / (1.0f + __expf(-s0[r] * scale));
        const float rel1 = 1.0f / (1.0f + __expf(-s1[r] * scale));
        rs0[r] += rel0;
        rs1[r] += rel1;
        ps0[(r + 8 * hi) * 32 + kb * 16 + cr] = (__bf16)rel0;  // C-frag -> LDS tile
        ps1[(r + 8 * hi) * 32 + kb * 16 + cr] = (__bf16)rel1;
      }
    }
    // P A-fragments from LDS (16x32, row stride 32)
    v16bf ap0 = load_a_frag(ps0 + cr * 32, hi);
    v16bf ap1 = load_a_frag(ps1 + cr * 32, hi);
    // V^T B-fragments: lane cr holds output column d = nb*16+cr; contiguous 32B
    const __bf16* vb = Vb + (size_t)cr * L + k32 + hi * 16;
    v16bf bv[4];
#pragma unroll
    for (int nb = 0; nb < 4; ++nb)
      bv[nb] = *(const v16bf*)(vb + (size_t)(16 * nb) * L);
#pragma unroll
    for (int nb = 0; nb < 4; ++nb) {
      o[nb]     = WMMA_BF16(ap0, bv[nb], o[nb]);
      o[4 + nb] = WMMA_BF16(ap1, bv[nb], o[4 + nb]);
    }
  }

  // Row-sum reduction across the 16 lanes of each half (wave32 shuffles)
  float inv0[8], inv1[8];
#pragma unroll
  for (int r = 0; r < 8; ++r) {
    float s = rs0[r];
    s += __shfl_xor(s, 1, 32); s += __shfl_xor(s, 2, 32);
    s += __shfl_xor(s, 4, 32); s += __shfl_xor(s, 8, 32);
    inv0[r] = 1.0f / fmaxf(s, 1.0f);
    float t = rs1[r];
    t += __shfl_xor(t, 1, 32); t += __shfl_xor(t, 2, 32);
    t += __shfl_xor(t, 4, 32); t += __shfl_xor(t, 8, 32);
    inv1[r] = 1.0f / fmaxf(t, 1.0f);
  }

  const int b = bh / Hn, h = bh - b * Hn;
  const int Dd = Hn * 64;
#pragma unroll
  for (int sub = 0; sub < 2; ++sub) {
#pragma unroll
    for (int nb = 0; nb < 4; ++nb) {
      v8f c = o[sub * 4 + nb];
      const int col = h * 64 + nb * 16 + cr;
#pragma unroll
      for (int r = 0; r < 8; ++r) {
        const int qr = q0 + sub * 16 + r + 8 * hi;
        const float iv = sub == 0 ? inv0[r] : inv1[r];
        att[((size_t)(b * L + qr)) * Dd + col] = (__bf16)(c[r] * iv);
      }
    }
  }
}

// ---------------------------------------------------------------------------
// Launch
// ---------------------------------------------------------------------------
extern "C" void kernel_launch(void* const* d_in, const int* in_sizes, int n_in,
                              void* d_out, int out_size, void* d_ws, size_t ws_size,
                              hipStream_t stream) {
  (void)in_sizes; (void)n_in; (void)out_size; (void)ws_size;
  constexpr int B = 4, L = 2048, D = 1024, H = 16;
  constexpr int M = B * L;              // 8192
  constexpr int K2 = 2 * D;             // 2048

  const float* z   = (const float*)d_in[0];
  const float* con = (const float*)d_in[1];
  const float* Wq  = (const float*)d_in[2];
  const float* Wk  = (const float*)d_in[3];
  const float* Wv  = (const float*)d_in[4];
  const float* Wo  = (const float*)d_in[5];
  const float* tmp = (const float*)d_in[6];

  char* w = (char*)d_ws;
  __bf16* zc  = (__bf16*)w; w += (size_t)M * K2 * 2;   // 32 MB
  __bf16* zb  = (__bf16*)w; w += (size_t)M * D * 2;    // 16 MB
  __bf16* WqT = (__bf16*)w; w += (size_t)D * K2 * 2;   //  4 MB
  __bf16* WkT = (__bf16*)w; w += (size_t)D * K2 * 2;   //  4 MB
  __bf16* WvT = (__bf16*)w; w += (size_t)D * D * 2;    //  2 MB
  __bf16* WoT = (__bf16*)w; w += (size_t)D * D * 2;    //  2 MB
  __bf16* Qh  = (__bf16*)w; w += (size_t)M * D * 2;    // 16 MB
  __bf16* Khd = (__bf16*)w; w += (size_t)M * D * 2;    // 16 MB
  __bf16* Vtb = (__bf16*)w; w += (size_t)M * D * 2;    // 16 MB
  __bf16* att = (__bf16*)w; w += (size_t)M * D * 2;    // 16 MB

  // 1) convert + concat inputs; transpose+convert weights
  build_inputs_bf16<<<(M * K2) / 256, 256, 0, stream>>>(z, con, zc, zb);
  transpose_cvt<<<(K2 * D + 255) / 256, 256, 0, stream>>>(Wq, WqT, K2, D);
  transpose_cvt<<<(K2 * D + 255) / 256, 256, 0, stream>>>(Wk, WkT, K2, D);
  transpose_cvt<<<(D * D + 255) / 256, 256, 0, stream>>>(Wv, WvT, D, D);
  transpose_cvt<<<(D * D + 255) / 256, 256, 0, stream>>>(Wo, WoT, D, D);

  // 2) projections: 8 waves/block, wave tile 32x64 -> blocks = (M/32)*(D/64)/8
  const int gemm_blocks = (M / 32) * (D / 64) / 8;    // 512
  gemm_bf16_wmma<<<gemm_blocks, 256, 0, stream>>>(zc, WqT, M, K2, D, 1, Qh,  L, H);
  gemm_bf16_wmma<<<gemm_blocks, 256, 0, stream>>>(zc, WkT, M, K2, D, 1, Khd, L, H);
  gemm_bf16_wmma<<<gemm_blocks, 256, 0, stream>>>(zb, WvT, M, D,  D, 2, Vtb, L, H);

  // 3) attention: one wave per (b,h, 32-row q-tile); 4 waves/block
  const int attn_waves = B * H * (L / 32);            // 4096
  attn_sigmoid_wmma<<<attn_waves / 4, 128, 0, stream>>>(Qh, Khd, Vtb, att, tmp,
                                                        L, H, B * H);

  // 4) output projection -> f32
  gemm_bf16_wmma<<<gemm_blocks, 256, 0, stream>>>(att, WoT, M, D, D, 0,
                                                  (float*)d_out, L, H);
}
